// APPNP_14370960572524
// MI455X (gfx1250) — matrix operand order repaired and checked by reading, not compiled, not run on hardware
//
#include <hip/hip_runtime.h>
#include <hip/hip_bf16.h>

typedef __bf16 bf16_t;
typedef __attribute__((ext_vector_type(16))) __bf16 v16bf;
typedef __attribute__((ext_vector_type(8)))  float  v8f;

#define IN_C  512
#define HID_C 256
#define OUT_C 16
#define K_ITERS 10
#define ALPHA_C 0.1f
#define BETA_C  0.9f

#define XSTR 520   // LDS row stride (bf16) for x tile: 512 + 8 pad
#define HSTR 264   // LDS row stride (bf16) for h1 tile: 256 + 8 pad

// ---------------- weight packing into WMMA B-fragment layout ----------------
// B (32x16 bf16) per-lane k order used here (contiguous, same as A): lanes 0-15 hold K=0..15 (N=lane),
// lanes 16-31 hold K=16..31 (N=lane-16).  Packed so each lane's 16 bf16
// fragment is a contiguous 32B run: pw[(kt*NT + nt)*512 + lane*16 + e]
__global__ void pack_w1_kernel(const float* __restrict__ W1, bf16_t* __restrict__ pw1) {
    int o = blockIdx.x * blockDim.x + threadIdx.x;          // 512*256 elements
    if (o >= IN_C * HID_C) return;
    int tile = o >> 9, lane = (o >> 4) & 31, e = o & 15;
    int kt = tile >> 4, nt = tile & 15;
    int k = kt * 32 + 16 * (lane >> 4) + e;
    int n = nt * 16 + (lane & 15);
    pw1[o] = (bf16_t)W1[k * HID_C + n];
}

__global__ void pack_w2_kernel(const float* __restrict__ W2, bf16_t* __restrict__ pw2) {
    int o = blockIdx.x * blockDim.x + threadIdx.x;          // 256*16 elements
    if (o >= HID_C * OUT_C) return;
    int kt = o >> 9, lane = (o >> 4) & 31, e = o & 15;
    int k = kt * 32 + 16 * (lane >> 4) + e;
    int n = lane & 15;
    pw2[o] = (bf16_t)W2[k * OUT_C + n];
}

// ---------------- fused 2-layer MLP: h0 = relu(x@W1+b1)@W2+b2 --------------
__global__ __launch_bounds__(256) void mlp_kernel(
    const float*  __restrict__ x,
    const bf16_t* __restrict__ pw1,
    const bf16_t* __restrict__ pw2,
    const float*  __restrict__ b1,
    const float*  __restrict__ b2,
    float*        __restrict__ h0,
    int n_nodes)
{
    __shared__ bf16_t lds_x[16 * XSTR];
    __shared__ bf16_t lds_h1[16 * HSTR];

    const int m0  = blockIdx.x * 16;
    const int tid = threadIdx.x;

    // stage x tile (16 x 512 f32 -> bf16) into LDS, shared by all 8 waves
    for (int i = tid; i < 16 * IN_C; i += 256) {
        int rr = i >> 9, cc = i & 511;
        int grow = m0 + rr;
        float v = (grow < n_nodes) ? x[(size_t)grow * IN_C + cc] : 0.0f;
        lds_x[rr * XSTR + cc] = (bf16_t)v;
    }
    __syncthreads();

    const int wave = tid >> 5;
    const int l    = tid & 31;
    const int half = l >> 4;
    const int r    = l & 15;       // A row / C column-lane
    const int col  = l & 15;

    // GEMM1: each wave owns 2 of 16 N-tiles (HID_C = 256)
    const int nt0 = wave * 2, nt1 = wave * 2 + 1;
    v8f acc0 = {}, acc1 = {};
    for (int kt = 0; kt < 16; ++kt) {
        v16bf a;
        const bf16_t* ap = &lds_x[r * XSTR + kt * 32 + 16 * half];
        #pragma unroll
        for (int e = 0; e < 16; ++e) { a[e] = ap[e]; }
        v16bf bf0, bf1;
        const bf16_t* bp0 = pw1 + ((size_t)(kt * 16 + nt0) * 512 + l * 16);
        const bf16_t* bp1 = pw1 + ((size_t)(kt * 16 + nt1) * 512 + l * 16);
        #pragma unroll
        for (int e = 0; e < 16; ++e) { bf0[e] = bp0[e]; bf1[e] = bp1[e]; }
        acc0 = __builtin_amdgcn_wmma_f32_16x16x32_bf16(false, a, false, bf0, (short)0, acc0, false, false);
        acc1 = __builtin_amdgcn_wmma_f32_16x16x32_bf16(false, a, false, bf1, (short)0, acc1, false, false);
    }

    // bias + relu, store h1 tile (16 x 256) to LDS as bf16
    const float bias0 = b1[nt0 * 16 + col];
    const float bias1 = b1[nt1 * 16 + col];
    #pragma unroll
    for (int g = 0; g < 8; ++g) {
        int mrow = g + 8 * half;                 // C/D layout: M = g + 8*half, N = lane&15
        float v0 = acc0[g] + bias0; v0 = v0 > 0.0f ? v0 : 0.0f;
        float v1 = acc1[g] + bias1; v1 = v1 > 0.0f ? v1 : 0.0f;
        lds_h1[mrow * HSTR + nt0 * 16 + col] = (bf16_t)v0;
        lds_h1[mrow * HSTR + nt1 * 16 + col] = (bf16_t)v1;
    }
    __syncthreads();

    // GEMM2: 16x256 @ 256x16, single N-tile -> wave 0 only (wave-uniform branch)
    if (wave == 0) {
        v8f acc2 = {};
        for (int kt = 0; kt < 8; ++kt) {
            v16bf a;
            const bf16_t* ap = &lds_h1[r * HSTR + kt * 32 + 16 * half];
            #pragma unroll
            for (int e = 0; e < 16; ++e) { a[e] = ap[e]; }
            v16bf bfr;
            const bf16_t* bp = pw2 + ((size_t)kt * 512 + l * 16);
            #pragma unroll
            for (int e = 0; e < 16; ++e) bfr[e] = bp[e];
            acc2 = __builtin_amdgcn_wmma_f32_16x16x32_bf16(false, a, false, bfr, (short)0, acc2, false, false);
        }
        const float bias2 = b2[col];
        #pragma unroll
        for (int g = 0; g < 8; ++g) {
            int mrow = g + 8 * half;
            int grow = m0 + mrow;
            if (grow < n_nodes)
                h0[(size_t)grow * OUT_C + col] = acc2[g] + bias2;
        }
    }
}

// ---------------- degree / normalization ----------------
__global__ void zero_deg_kernel(float* __restrict__ deg, int n) {
    int i = blockIdx.x * blockDim.x + threadIdx.x;
    if (i < n) deg[i] = 0.0f;
}

__global__ void deg_edge_kernel(const int* __restrict__ dst, float* __restrict__ deg, int E) {
    int e = blockIdx.x * blockDim.x + threadIdx.x;
    if (e < E) atomicAdd(&deg[dst[e]], 1.0f);
}

// self-loop contributes +1 to every node's degree; deg -> deg^{-1/2} in place
__global__ void dinv_kernel(float* __restrict__ deg, int n) {
    int i = blockIdx.x * blockDim.x + threadIdx.x;
    if (i < n) deg[i] = rsqrtf(deg[i] + 1.0f);
}

// ---------------- APPNP propagation ----------------
// hnext = beta * dinv[i]^2 * hcur[i]  (self-loop edge)  +  alpha * h0[i]
__global__ void appnp_init_kernel(const float* __restrict__ hcur,
                                  const float* __restrict__ h0,
                                  const float* __restrict__ dinv,
                                  float* __restrict__ hnext, int n) {
    int i = blockIdx.x * blockDim.x + threadIdx.x;
    int node = i >> 4;
    if (node >= n) return;
    float di = dinv[node];
    hnext[i] = BETA_C * di * di * hcur[i] + ALPHA_C * h0[i];
}

__global__ void appnp_edge_kernel(const int* __restrict__ src,
                                  const int* __restrict__ dst,
                                  const float* __restrict__ dinv,
                                  const float* __restrict__ hcur,
                                  float* __restrict__ hnext, int E) {
    int e = blockIdx.x * blockDim.x + threadIdx.x;
    if (e >= E) return;
    int s = src[e], d = dst[e];
    float w = BETA_C * dinv[s] * dinv[d];
    const float4* hs = (const float4*)(hcur + (size_t)s * OUT_C);
    float4 q0 = hs[0], q1 = hs[1], q2 = hs[2], q3 = hs[3];
    float* hd = hnext + (size_t)d * OUT_C;
    atomicAdd(hd + 0,  w * q0.x); atomicAdd(hd + 1,  w * q0.y);
    atomicAdd(hd + 2,  w * q0.z); atomicAdd(hd + 3,  w * q0.w);
    atomicAdd(hd + 4,  w * q1.x); atomicAdd(hd + 5,  w * q1.y);
    atomicAdd(hd + 6,  w * q1.z); atomicAdd(hd + 7,  w * q1.w);
    atomicAdd(hd + 8,  w * q2.x); atomicAdd(hd + 9,  w * q2.y);
    atomicAdd(hd + 10, w * q2.z); atomicAdd(hd + 11, w * q2.w);
    atomicAdd(hd + 12, w * q3.x); atomicAdd(hd + 13, w * q3.y);
    atomicAdd(hd + 14, w * q3.z); atomicAdd(hd + 15, w * q3.w);
}

// ---------------- log-softmax over 16 channels ----------------
__global__ void logsoftmax_kernel(const float* __restrict__ h, float* __restrict__ out, int n) {
    int node = blockIdx.x * blockDim.x + threadIdx.x;
    if (node >= n) return;
    const float4* p = (const float4*)(h + (size_t)node * OUT_C);
    float v[16];
    float4 q0 = p[0], q1 = p[1], q2 = p[2], q3 = p[3];
    v[0]=q0.x; v[1]=q0.y; v[2]=q0.z; v[3]=q0.w;
    v[4]=q1.x; v[5]=q1.y; v[6]=q1.z; v[7]=q1.w;
    v[8]=q2.x; v[9]=q2.y; v[10]=q2.z; v[11]=q2.w;
    v[12]=q3.x; v[13]=q3.y; v[14]=q3.z; v[15]=q3.w;
    float m = v[0];
    #pragma unroll
    for (int i = 1; i < 16; ++i) m = v[i] > m ? v[i] : m;
    float s = 0.0f;
    #pragma unroll
    for (int i = 0; i < 16; ++i) s += __expf(v[i] - m);
    float ls = __logf(s);
    float* o = out + (size_t)node * OUT_C;
    #pragma unroll
    for (int i = 0; i < 16; ++i) o[i] = v[i] - m - ls;
}

// ---------------- launcher ----------------
extern "C" void kernel_launch(void* const* d_in, const int* in_sizes, int n_in,
                              void* d_out, int out_size, void* d_ws, size_t ws_size,
                              hipStream_t stream) {
    const float* x  = (const float*)d_in[0];
    const int*   ei = (const int*)  d_in[1];
    const float* W1 = (const float*)d_in[2];
    const float* b1 = (const float*)d_in[3];
    const float* W2 = (const float*)d_in[4];
    const float* b2 = (const float*)d_in[5];
    float* out = (float*)d_out;

    const int N = in_sizes[0] / IN_C;     // 100000
    const int E = in_sizes[1] / 2;        // 3200000
    const int* src = ei;
    const int* dst = ei + E;

    // workspace carve-out (256B aligned)
    char* ws = (char*)d_ws;
    size_t off = 0;
    auto carve = [&](size_t bytes) -> void* {
        void* p = (void*)(ws + off);
        off = (off + bytes + 255) & ~(size_t)255;
        return p;
    };
    bf16_t* pw1  = (bf16_t*)carve((size_t)IN_C * HID_C * sizeof(bf16_t));
    bf16_t* pw2  = (bf16_t*)carve((size_t)HID_C * OUT_C * sizeof(bf16_t));
    float*  dinv = (float*) carve((size_t)N * sizeof(float));
    float*  h0   = (float*) carve((size_t)N * OUT_C * sizeof(float));
    float*  hA   = (float*) carve((size_t)N * OUT_C * sizeof(float));
    float*  hB   = (float*) carve((size_t)N * OUT_C * sizeof(float));

    // 1) pack weights into WMMA B-fragment layout (bf16)
    pack_w1_kernel<<<(IN_C * HID_C + 255) / 256, 256, 0, stream>>>(W1, pw1);
    pack_w2_kernel<<<(HID_C * OUT_C + 255) / 256, 256, 0, stream>>>(W2, pw2);

    // 2) degrees -> dinv = (deg+1)^{-1/2}
    zero_deg_kernel<<<(N + 255) / 256, 256, 0, stream>>>(dinv, N);
    deg_edge_kernel<<<(E + 255) / 256, 256, 0, stream>>>(dst, dinv, E);
    dinv_kernel<<<(N + 255) / 256, 256, 0, stream>>>(dinv, N);

    // 3) fused MLP via WMMA bf16
    mlp_kernel<<<(N + 15) / 16, 256, 0, stream>>>(x, pw1, pw2, b1, b2, h0, N);

    // 4) APPNP: K rounds of propagate
    const float* cur = h0;
    float* nxt = hA;
    for (int k = 0; k < K_ITERS; ++k) {
        appnp_init_kernel<<<((size_t)N * OUT_C + 255) / 256, 256, 0, stream>>>(cur, h0, dinv, nxt, N);
        appnp_edge_kernel<<<(E + 255) / 256, 256, 0, stream>>>(src, dst, dinv, cur, nxt, E);
        cur = nxt;
        nxt = (cur == hA) ? hB : hA;
    }

    // 5) log-softmax
    logsoftmax_kernel<<<(N + 255) / 256, 256, 0, stream>>>(cur, out, N);
}
